// Encoder_overall_23768349016376
// MI455X (gfx1250) — compile-verified, matrix operand order checked by reading, not run on hardware
//
#include <hip/hip_runtime.h>
#include <hip/hip_bf16.h>
#include <math.h>

// ---------------------------------------------------------------------------
// Encoder_overall for MI455X (gfx1250, wave32, WMMA).
// f32 streamed from HBM, converted to bf16 in registers, accumulated with
// v_wmma_f32_16x16x32_bf16 in f32. recon refactored: Adj@(Z@W) == (Adj@Z)@W.
// K-loop split main/tail (branch-free hot loop), NOUT is a template constant
// so all B-fragment offsets fold into immediate load offsets.
// ---------------------------------------------------------------------------

#define NSPOTS 30000

typedef __attribute__((ext_vector_type(16))) __bf16 bf16x16;
typedef __attribute__((ext_vector_type(8)))  float  f32x8;

// K-offset of element q (0..15) of a v16bf A/B fragment, per ISA 7.12.2
// 16-bit 16x32 layout: lanes<16 hold K {0..7,16..23}, lanes>=16 hold {8..15,24..31}.
__device__ __forceinline__ int koff(int q, int kgrp) {
  return (q & 7) | (kgrp << 3) | ((q & 8) << 1);
}

__device__ __forceinline__ float act_apply(float v, int act) {
  if (act == 1)      v = fmaxf(v, 0.0f);                 // relu
  else if (act == 2) v = 1.0f / (1.0f + __expf(-v));     // sigmoid
  else if (act == 3) v = tanhf(v);                       // tanh
  return v;
}

// Fast A fragment: two 32B f32 segments per lane (K offsets base..+7, base+16..+23)
__device__ __forceinline__ bf16x16 load_a_fast(const float* xr, int kb, int kgrp) {
  union { bf16x16 v; __bf16 e[16]; } a;
  const float4* p = (const float4*)(xr + kb + (kgrp << 3));
  float4 f0 = p[0], f1 = p[1], f2 = p[4], f3 = p[5];
  a.e[0]=(__bf16)f0.x;  a.e[1]=(__bf16)f0.y;  a.e[2]=(__bf16)f0.z;  a.e[3]=(__bf16)f0.w;
  a.e[4]=(__bf16)f1.x;  a.e[5]=(__bf16)f1.y;  a.e[6]=(__bf16)f1.z;  a.e[7]=(__bf16)f1.w;
  a.e[8]=(__bf16)f2.x;  a.e[9]=(__bf16)f2.y;  a.e[10]=(__bf16)f2.z; a.e[11]=(__bf16)f2.w;
  a.e[12]=(__bf16)f3.x; a.e[13]=(__bf16)f3.y; a.e[14]=(__bf16)f3.z; a.e[15]=(__bf16)f3.w;
  return a.v;
}

// ---------------------------------------------------------------------------
// gemm_kf<NOUT>: out[M x NOUT] = act(X[M x K] @ W[K x NOUT] + bias)
// One wave: 16 rows x NOUT cols. NOUT compile-time => B offsets are immediates
// and all column guards vanish. If X2 != nullptr, logical row r reads
// (r odd ? X2 : X)[r>>1]  (stack([e1,e2],axis=1)).
// ---------------------------------------------------------------------------
template<int NOUT>
__global__ __launch_bounds__(256) void gemm_kf(
    const float* __restrict__ X, const float* __restrict__ X2,
    const float* __restrict__ W, const float* __restrict__ bias,
    float* __restrict__ out, int M, int K, int act) {
  constexpr int NT = (NOUT + 15) / 16;
  const int lane = threadIdx.x & 31;
  const int wid  = threadIdx.x >> 5;
  const int rowTile = blockIdx.x * (blockDim.x >> 5) + wid;
  const int r0 = rowTile << 4;
  if (r0 >= M) return;
  const int kgrp = lane >> 4;
  const int mrow = lane & 15;
  const int row  = r0 + mrow;
  const float* xr = X2 ? (((row & 1) ? X2 : X) + (size_t)(row >> 1) * K)
                       : (X + (size_t)row * K);
  const int ncol = lane & 15;

  f32x8 c[NT] = {};
  const int Kmain = K & ~31;

  // ---- branch-free main K loop ----
  for (int kb = 0; kb < Kmain; kb += 32) {
    const bf16x16 av = load_a_fast(xr, kb, kgrp);
    // base of this K-chunk for this lane's kgrp half
    const float* wb = W + (size_t)(kb + (kgrp << 3)) * NOUT + ncol;
    #pragma unroll
    for (int t = 0; t < NT; ++t) {
      constexpr bool tileFull = true;  // guards handled below for NOUT<16
      union { bf16x16 v; __bf16 e[16]; } b;
      if constexpr ((NT * 16) == NOUT) {
        // koff(q,kgrp) = kgrp*8 + (q&7) + ((q&8)<<1): fold into immediates
        #pragma unroll
        for (int q = 0; q < 16; ++q)
          b.e[q] = (__bf16)wb[(size_t)(((q & 7) | ((q & 8) << 1))) * NOUT + (t << 4)];
      } else {
        const int n = (t << 4) + ncol;
        const int nn = (n < NOUT) ? n : (NOUT - 1);
        #pragma unroll
        for (int q = 0; q < 16; ++q) {
          const int k = kb + koff(q, kgrp);
          const float x = W[(size_t)k * NOUT + nn];
          b.e[q] = (__bf16)((n < NOUT) ? x : 0.0f);
        }
      }
      (void)tileFull;
      c[t] = __builtin_amdgcn_wmma_f32_16x16x32_bf16(
                 false, av, false, b.v, (short)0, c[t], false, false);
    }
  }

  // ---- single K tail chunk (branchless clamped loads + selects) ----
  if (Kmain < K) {
    const int kb = Kmain;
    union { bf16x16 v; __bf16 e[16]; } a;
    #pragma unroll
    for (int q = 0; q < 16; ++q) {
      const int k  = kb + koff(q, kgrp);
      const int kk = (k < K) ? k : (K - 1);
      const float x = xr[kk];
      a.e[q] = (__bf16)((k < K) ? x : 0.0f);
    }
    #pragma unroll
    for (int t = 0; t < NT; ++t) {
      const int n  = (t << 4) + ncol;
      const int nn = (n < NOUT) ? n : (NOUT - 1);
      union { bf16x16 v; __bf16 e[16]; } b;
      #pragma unroll
      for (int q = 0; q < 16; ++q) {
        const int k  = kb + koff(q, kgrp);
        const int kk = (k < K) ? k : (K - 1);
        const float x = W[(size_t)kk * NOUT + nn];
        b.e[q] = (__bf16)(((k < K) && (n < NOUT)) ? x : 0.0f);
      }
      c[t] = __builtin_amdgcn_wmma_f32_16x16x32_bf16(
                 false, a.v, false, b.v, (short)0, c[t], false, false);
    }
  }

  // ---- epilogue: bias + activation + store ----
  #pragma unroll
  for (int t = 0; t < NT; ++t) {
    const int n = (t << 4) + ncol;
    if constexpr ((NT * 16) != NOUT) {
      if (n >= NOUT) continue;
    }
    const float bs = bias ? bias[n] : 0.0f;
    #pragma unroll
    for (int i = 0; i < 8; ++i) {
      const int r = r0 + i + (kgrp << 3);
      out[(size_t)r * NOUT + n] = act_apply(c[t][i] + bs, act);
    }
  }
}

// ---------------------------------------------------------------------------
// gemm_nf<KC>: out[M x Nout] = X[M x 32*KC] @ W[32*KC x Nout]
// A fragments preloaded once per wave, branch-free sweep over full N-tiles,
// one ragged tile at the end (wide recon GEMMs: K=64, Nout=3000/1000).
// ---------------------------------------------------------------------------
template<int KC>
__global__ __launch_bounds__(256) void gemm_nf(
    const float* __restrict__ X, const float* __restrict__ W,
    float* __restrict__ out, int M, int Nout, int act) {
  const int K = KC * 32;
  const int lane = threadIdx.x & 31;
  const int wid  = threadIdx.x >> 5;
  const int rowTile = blockIdx.x * (blockDim.x >> 5) + wid;
  const int r0 = rowTile << 4;
  if (r0 >= M) return;
  const int kgrp = lane >> 4;
  const int mrow = lane & 15;
  const float* xr = X + (size_t)(r0 + mrow) * K;
  const int ncol = lane & 15;

  bf16x16 a[KC];
  #pragma unroll
  for (int kc = 0; kc < KC; ++kc)
    a[kc] = load_a_fast(xr, kc << 5, kgrp);

  const int Nfull = Nout & ~15;

  // ---- branch-free main N loop ----
  for (int n0 = 0; n0 < Nfull; n0 += 16) {
    const int n = n0 + ncol;
    f32x8 c = {};
    #pragma unroll
    for (int kc = 0; kc < KC; ++kc) {
      union { bf16x16 v; __bf16 e[16]; } b;
      #pragma unroll
      for (int q = 0; q < 16; ++q)
        b.e[q] = (__bf16)W[(size_t)((kc << 5) + koff(q, kgrp)) * Nout + n];
      c = __builtin_amdgcn_wmma_f32_16x16x32_bf16(
              false, a[kc], false, b.v, (short)0, c, false, false);
    }
    #pragma unroll
    for (int i = 0; i < 8; ++i)
      out[(size_t)(r0 + i + (kgrp << 3)) * Nout + n] = act_apply(c[i], act);
  }

  // ---- one ragged N tile ----
  if (Nfull < Nout) {
    const int n  = Nfull + ncol;
    const int nn = (n < Nout) ? n : (Nout - 1);
    f32x8 c = {};
    #pragma unroll
    for (int kc = 0; kc < KC; ++kc) {
      union { bf16x16 v; __bf16 e[16]; } b;
      #pragma unroll
      for (int q = 0; q < 16; ++q) {
        const float x = W[(size_t)((kc << 5) + koff(q, kgrp)) * Nout + nn];
        b.e[q] = (__bf16)((n < Nout) ? x : 0.0f);
      }
      c = __builtin_amdgcn_wmma_f32_16x16x32_bf16(
              false, a[kc], false, b.v, (short)0, c, false, false);
    }
    if (n < Nout) {
      #pragma unroll
      for (int i = 0; i < 8; ++i)
        out[(size_t)(r0 + i + (kgrp << 3)) * Nout + n] = act_apply(c[i], act);
    }
  }
}

// ---------------------------------------------------------------------------
// spmm: out[rows[e]] += vals[e] * z[cols[e]]  (z,out are N x 64; out pre-zeroed)
// rows sorted: each wave walks a contiguous edge range with a segmented
// accumulator (2 dims/lane). Edge metadata forced into SGPRs so the segment
// boundary is a scalar branch; flushes use global_atomic_add_f32.
// ---------------------------------------------------------------------------
__global__ __launch_bounds__(256) void spmm_kernel(
    const int* __restrict__ rows, const int* __restrict__ cols,
    const float* __restrict__ vals, const float* __restrict__ z,
    float* __restrict__ out, int E, int edgesPerWave) {
  const int lane = threadIdx.x & 31;
  const long wave = (long)((blockIdx.x * blockDim.x + threadIdx.x) >> 5);
  long e0 = wave * edgesPerWave;
  if (e0 >= E) return;
  long e1 = e0 + edgesPerWave;
  if (e1 > E) e1 = E;

  float a0 = 0.0f, a1 = 0.0f;
  int cur = __builtin_amdgcn_readfirstlane(rows[e0]);
  for (long e = e0; e < e1; ++e) {
    const int r  = __builtin_amdgcn_readfirstlane(rows[e]);
    const int cc = __builtin_amdgcn_readfirstlane(cols[e]);
    const float v = __uint_as_float(
        __builtin_amdgcn_readfirstlane(__float_as_uint(vals[e])));
    if (r != cur) {                               // scalar branch (uniform)
      atomicAdd(&out[(size_t)cur * 64 + lane], a0);
      atomicAdd(&out[(size_t)cur * 64 + lane + 32], a1);
      cur = r; a0 = 0.0f; a1 = 0.0f;
    }
    a0 = fmaf(v, z[(size_t)cc * 64 + lane], a0);
    a1 = fmaf(v, z[(size_t)cc * 64 + lane + 32], a1);
  }
  atomicAdd(&out[(size_t)cur * 64 + lane], a0);
  atomicAdd(&out[(size_t)cur * 64 + lane + 32], a1);
}

// ---------------------------------------------------------------------------
// att_combine: given v = tanh(stack([e1,e2]) @ W) (precomputed, [2N x 64]),
// compute vu = v @ u, 2-way softmax (+1e-6), blend e1/e2. One wave per node.
// ---------------------------------------------------------------------------
__global__ __launch_bounds__(256) void att_combine(
    const float* __restrict__ e1, const float* __restrict__ e2,
    const float* __restrict__ v, const float* __restrict__ u,
    float* __restrict__ outc, int N) {
  const int lane = threadIdx.x & 31;
  const int wave = (blockIdx.x * blockDim.x + threadIdx.x) >> 5;
  if (wave >= N) return;
  const size_t n = (size_t)wave;

  const float u0 = u[lane], u1 = u[lane + 32];
  const float* v1 = v + (2 * n) * 64;
  const float* v2 = v1 + 64;
  float s1 = v1[lane] * u0 + v1[lane + 32] * u1;
  float s2 = v2[lane] * u0 + v2[lane + 32] * u1;
  #pragma unroll
  for (int off = 16; off > 0; off >>= 1) {
    s1 += __shfl_xor(s1, off, 32);
    s2 += __shfl_xor(s2, off, 32);
  }
  s1 += 1e-6f; s2 += 1e-6f;
  const float m  = fmaxf(s1, s2);
  const float x1 = __expf(s1 - m), x2 = __expf(s2 - m);
  const float inv = 1.0f / (x1 + x2);
  const float al1 = x1 * inv, al2 = x2 * inv;

  const float* p1 = e1 + n * 64;
  const float* p2 = e2 + n * 64;
  float* o = outc + n * 64;
  o[lane]      = al1 * p1[lane]      + al2 * p2[lane];
  o[lane + 32] = al1 * p1[lane + 32] + al2 * p2[lane + 32];
}

__global__ void zero_f32(float* __restrict__ p, size_t n) {
  size_t i = (size_t)blockIdx.x * blockDim.x + threadIdx.x;
  const size_t stride = (size_t)gridDim.x * blockDim.x;
  for (; i < n; i += stride) p[i] = 0.0f;
}

// ---------------------------------------------------------------------------
// Host-side orchestration
// ---------------------------------------------------------------------------
static inline int kf_blocks(int M) { return ((M + 15) / 16 + 7) / 8; }

extern "C" void kernel_launch(void* const* d_in, const int* in_sizes, int n_in,
                              void* d_out, int out_size, void* d_ws, size_t ws_size,
                              hipStream_t stream) {
  const int N = NSPOTS, IN1 = 3000, IN2 = 1000;
  const int E = in_sizes[2];

  const float* feat1  = (const float*)d_in[0];
  const float* feat2  = (const float*)d_in[1];
  const int*   sp1_r  = (const int*)d_in[2];
  const int*   sp1_c  = (const int*)d_in[3];
  const float* sp1_v  = (const float*)d_in[4];
  const int*   ft1_r  = (const int*)d_in[5];
  const int*   ft1_c  = (const int*)d_in[6];
  const float* ft1_v  = (const float*)d_in[7];
  const int*   sp2_r  = (const int*)d_in[8];
  const int*   sp2_c  = (const int*)d_in[9];
  const float* sp2_v  = (const float*)d_in[10];
  const int*   ft2_r  = (const int*)d_in[11];
  const int*   ft2_c  = (const int*)d_in[12];
  const float* ft2_v  = (const float*)d_in[13];
  const float* enc1_w = (const float*)d_in[14];
  const float* enc2_w = (const float*)d_in[15];
  const float* dec1_w = (const float*)d_in[16];
  const float* dec2_w = (const float*)d_in[17];
  const float* att1_w = (const float*)d_in[18];
  const float* att1_u = (const float*)d_in[19];
  const float* att2_w = (const float*)d_in[20];
  const float* att2_u = (const float*)d_in[21];
  const float* attc_w = (const float*)d_in[22];
  const float* attc_u = (const float*)d_in[23];
  const float* t12_w1 = (const float*)d_in[24]; const float* t12_b1 = (const float*)d_in[25];
  const float* t12_w2 = (const float*)d_in[26]; const float* t12_b2 = (const float*)d_in[27];
  const float* t12_w3 = (const float*)d_in[28]; const float* t12_b3 = (const float*)d_in[29];
  const float* t21_w1 = (const float*)d_in[30]; const float* t21_b1 = (const float*)d_in[31];
  const float* t21_w2 = (const float*)d_in[32]; const float* t21_b2 = (const float*)d_in[33];
  const float* t21_w3 = (const float*)d_in[34]; const float* t21_b3 = (const float*)d_in[35];
  const float* d1_w1  = (const float*)d_in[36]; const float* d1_b1  = (const float*)d_in[37];
  const float* d1_w2  = (const float*)d_in[38]; const float* d1_b2  = (const float*)d_in[39];
  const float* d1_w3  = (const float*)d_in[40]; const float* d1_b3  = (const float*)d_in[41];
  const float* d2_w1  = (const float*)d_in[42]; const float* d2_b1  = (const float*)d_in[43];
  const float* d2_w2  = (const float*)d_in[44]; const float* d2_b2  = (const float*)d_in[45];
  const float* d2_w3  = (const float*)d_in[46]; const float* d2_b3  = (const float*)d_in[47];

  // Output layout (concatenated in return order)
  const size_t U = (size_t)N * 64;
  float* out      = (float*)d_out;
  float* emb1     = out;                 // N*64
  float* emb2     = emb1 + U;
  float* e12      = emb2 + U;
  float* e21      = e12 + U;
  float* pred1    = e21 + U;             // N
  float* pred2    = pred1 + N;
  float* embc     = pred2 + N;           // N*64
  float* recon1   = embc + U;            // N*IN1
  float* recon2   = recon1 + (size_t)N * IN1;

  // Workspace layout (units of N*64 floats)
  float* w  = (float*)d_ws;
  float* u0 = w;            // z1, later g1 = spmm(sp1, emb_combined)
  float* u1 = w + 1 * U;    // z2, later g2
  float* u2 = w + 2 * U;    // emb_sp1
  float* u3 = w + 3 * U;    // emb_ft1
  float* u4 = w + 4 * U;    // emb_sp2
  float* u5 = w + 5 * U;    // emb_ft2
  float* u6 = w + 6 * U;    // vbuf [2N x 64]  /  h128 [N x 128]  (2 units)
  float* u8 = w + 8 * U;    // h64

  const int epw = 128;                              // edges per wave (spmm)
  const int spmmBlocks = ((E + epw - 1) / epw + 7) / 8;
  const int attBlocks  = (N + 7) / 8;

  // 1) GCN encoders: z = feat @ enc_w
  gemm_kf<64><<<kf_blocks(N), 256, 0, stream>>>(feat1, nullptr, enc1_w, nullptr, u0, N, IN1, 0);
  gemm_kf<64><<<kf_blocks(N), 256, 0, stream>>>(feat2, nullptr, enc2_w, nullptr, u1, N, IN2, 0);

  // 2) SpMMs into zeroed buffers
  zero_f32<<<1024, 256, 0, stream>>>(u2, 4 * U);
  spmm_kernel<<<spmmBlocks, 256, 0, stream>>>(sp1_r, sp1_c, sp1_v, u0, u2, E, epw);
  spmm_kernel<<<spmmBlocks, 256, 0, stream>>>(ft1_r, ft1_c, ft1_v, u0, u3, E, epw);
  spmm_kernel<<<spmmBlocks, 256, 0, stream>>>(sp2_r, sp2_c, sp2_v, u1, u4, E, epw);
  spmm_kernel<<<spmmBlocks, 256, 0, stream>>>(ft2_r, ft2_c, ft2_v, u1, u5, E, epw);

  // 3) Attention 1 & 2: v = tanh(stack(e_sp,e_ft) @ W), combine
  gemm_kf<64><<<kf_blocks(2 * N), 256, 0, stream>>>(u2, u3, att1_w, nullptr, u6, 2 * N, 64, 3);
  att_combine<<<attBlocks, 256, 0, stream>>>(u2, u3, u6, att1_u, emb1, N);
  gemm_kf<64><<<kf_blocks(2 * N), 256, 0, stream>>>(u4, u5, att2_w, nullptr, u6, 2 * N, 64, 3);
  att_combine<<<attBlocks, 256, 0, stream>>>(u4, u5, u6, att2_u, emb2, N);

  // 4) Combined attention (before u6 is recycled as MLP hidden)
  gemm_kf<64><<<kf_blocks(2 * N), 256, 0, stream>>>(emb1, emb2, attc_w, nullptr, u6, 2 * N, 64, 3);
  att_combine<<<attBlocks, 256, 0, stream>>>(emb1, emb2, u6, attc_u, embc, N);

  // 5) Translation MLPs
  gemm_kf<128><<<kf_blocks(N), 256, 0, stream>>>(emb1, nullptr, t12_w1, t12_b1, u6, N, 64, 1);
  gemm_kf<64><<<kf_blocks(N), 256, 0, stream>>>(u6,   nullptr, t12_w2, t12_b2, u8, N, 128, 1);
  gemm_kf<64><<<kf_blocks(N), 256, 0, stream>>>(u8,   nullptr, t12_w3, t12_b3, e12, N, 64, 0);
  gemm_kf<128><<<kf_blocks(N), 256, 0, stream>>>(emb2, nullptr, t21_w1, t21_b1, u6, N, 64, 1);
  gemm_kf<64><<<kf_blocks(N), 256, 0, stream>>>(u6,   nullptr, t21_w2, t21_b2, u8, N, 128, 1);
  gemm_kf<64><<<kf_blocks(N), 256, 0, stream>>>(u8,   nullptr, t21_w3, t21_b3, e21, N, 64, 0);

  // 6) Discriminator MLPs (final sigmoid, Nout=1)
  gemm_kf<128><<<kf_blocks(N), 256, 0, stream>>>(emb1, nullptr, d1_w1, d1_b1, u6, N, 64, 1);
  gemm_kf<64><<<kf_blocks(N), 256, 0, stream>>>(u6,   nullptr, d1_w2, d1_b2, u8, N, 128, 1);
  gemm_kf<1><<<kf_blocks(N), 256, 0, stream>>>(u8,    nullptr, d1_w3, d1_b3, pred1, N, 64, 2);
  gemm_kf<128><<<kf_blocks(N), 256, 0, stream>>>(emb2, nullptr, d2_w1, d2_b1, u6, N, 64, 1);
  gemm_kf<64><<<kf_blocks(N), 256, 0, stream>>>(u6,   nullptr, d2_w2, d2_b2, u8, N, 128, 1);
  gemm_kf<1><<<kf_blocks(N), 256, 0, stream>>>(u8,    nullptr, d2_w3, d2_b3, pred2, N, 64, 2);

  // 7) Recon: Adj@(Z@W) == (Adj@Z)@W -> spmm first (64-dim), then wide GEMM
  zero_f32<<<1024, 256, 0, stream>>>(u0, 2 * U);
  spmm_kernel<<<spmmBlocks, 256, 0, stream>>>(sp1_r, sp1_c, sp1_v, embc, u0, E, epw);
  spmm_kernel<<<spmmBlocks, 256, 0, stream>>>(sp2_r, sp2_c, sp2_v, embc, u1, E, epw);
  gemm_nf<2><<<kf_blocks(N), 256, 0, stream>>>(u0, dec1_w, recon1, N, IN1, 0);
  gemm_nf<2><<<kf_blocks(N), 256, 0, stream>>>(u1, dec2_w, recon2, N, IN2, 0);
}
